// Convolution_20315195310850
// MI455X (gfx1250) — matrix operand order repaired and checked by reading, not compiled, not run on hardware
//
#include <hip/hip_runtime.h>
#include <hip/hip_bf16.h>
#include <math.h>

// CDNA5 / gfx1250, wave32. Implicit-GEMM 3D conv via V_WMMA_F32_16X16X32_BF16
// using split-bf16 (hi/lo) operands for ~fp32 accuracy on the fast pipe.
// Self-connection folded into the center tap of the conv filter.
// Filter slices staged into LDS with GLOBAL_LOAD_ASYNC_TO_LDS_B128 (ASYNCcnt).

typedef __attribute__((ext_vector_type(16))) __bf16 v16bf;
typedef __attribute__((ext_vector_type(8)))  float  v8f;

#define NX  64
#define CHW (NX * NX * NX)   // per-channel voxels
#define NCH 32

// fragment geometry (bf16 A-fragments, 16 data + 8 pad ushorts per lane)
#define LANE_STRIDE 24                    // ushorts per lane (48 B, 16B aligned)
#define FRAG_STRIDE (32 * LANE_STRIDE)    // 768 ushorts per fragment
#define TAP_STRIDE  (4 * FRAG_STRIDE)     // 3072 ushorts per tap (hi/lo x mt0/1)
#define BROW        40                    // input-plane LDS row stride (ushorts)

__device__ __forceinline__ unsigned short f2bf(float f) {
  unsigned u = __builtin_bit_cast(unsigned, f);
  u = (u + 0x7FFFu + ((u >> 16) & 1u)) >> 16;     // RNE
  return (unsigned short)u;
}
__device__ __forceinline__ float bf2f(unsigned short h) {
  return __builtin_bit_cast(float, (unsigned)h << 16);
}

#define WMMA_BF16(A, B, C) __builtin_amdgcn_wmma_f32_16x16x32_bf16( \
    false, (A), false, (B), (short)0, (C), false, false)

// ---------------------------------------------------------------------------
// Kernel 1: build the 125 x 32 x 32 filter (sc folded into tap (2,2,2)) and
// store hi/lo bf16 A-fragments:
//   frag f = mt*2 + plane (plane 0 = hi, 1 = lo), per-lane 16 bf16:
//   element e -> K = (lane>>4)*16 + e, M = mt*16 + (lane&15)
//   A[M][K] = kern[t, i=K, o=M]
// Tap block = 6144 B; the 5-tap (dz) slice per (dx,dy) is contiguous 30 KB.
// Total ws: 125 * 6144 = 750 KB.
// ---------------------------------------------------------------------------
__global__ __launch_bounds__(128)
void build_filter(const float* __restrict__ weight,   // (5, 320)
                  const float* __restrict__ wsc0,     // (8, 8)
                  const float* __restrict__ wsc1,     // (8, 8)
                  unsigned short* __restrict__ filtb) {
  __shared__ float ks[32][33];
  const int t  = blockIdx.x;                 // tap 0..124
  const int dx = t / 25, dy = (t / 5) % 5, dz = t % 5;
  const float ci = (float)(dx - 2), cj = (float)(dy - 2), ck = (float)(dz - 2);
  const float d  = sqrtf(ci * ci + cj * cj + ck * ck);

  // soft radial basis: emb[n] = C * sus(d/step - n) * sus(n+2 - d/step)
  const float step = 2.5f / 6.0f;
  const float C    = 1.14136f * expf(2.0f);
  const float r    = d / step;
  float emb[5];
#pragma unroll
  for (int n = 0; n < 5; ++n) {
    float t0 = r - (float)n;
    float t1 = (float)(n + 2) - r;
    float s0 = (t0 > 0.f) ? expf(-1.f / t0) : 0.f;
    float s1 = (t1 > 0.f) ? expf(-1.f / t1) : 0.f;
    emb[n] = C * s0 * s1;
  }

  // real spherical harmonics of unit offset
  float ux = 0.f, uy = 0.f, uz = 0.f;
  if (d > 0.f) { ux = ci / d; uy = cj / d; uz = ck / d; }
  const float s3 = 1.7320508075688772f, s5 = 2.23606797749979f,
              s15 = 3.872983346207417f;
  float sh[9];
  sh[0] = 1.f;
  sh[1] = s3 * ux; sh[2] = s3 * uy; sh[3] = s3 * uz;
  sh[4] = s15 * ux * uz;
  sh[5] = s15 * ux * uy;
  sh[6] = s5 * (uy * uy - 0.5f * (ux * ux + uz * uz));
  sh[7] = s15 * uy * uz;
  sh[8] = 0.5f * s15 * (uz * uz - ux * ux);

  // T[m][k] = (1/sqrt5) * sum_j sh2[j] * S[j][m][k]
  const float i_s2 = 0.7071067811865475f;
  const float i_s6 = 0.4082482904638631f;
  const float i_s5 = 0.4472135954999579f;
  float T[3][3];
  T[0][0] = i_s5 * (-sh[6] * i_s6 - sh[8] * i_s2);
  T[0][1] = i_s5 * ( sh[5] * i_s2);
  T[0][2] = i_s5 * ( sh[4] * i_s2);
  T[1][0] = T[0][1];
  T[1][1] = i_s5 * ( 2.f * sh[6] * i_s6);
  T[1][2] = i_s5 * ( sh[7] * i_s2);
  T[2][0] = T[0][2];
  T[2][1] = T[1][2];
  T[2][2] = i_s5 * (-sh[6] * i_s6 + sh[8] * i_s2);

  auto Wcol = [&](int c) -> float {
    float s = 0.f;
#pragma unroll
    for (int n = 0; n < 5; ++n) s += emb[n] * weight[n * 320 + c];
    return s * (1.0f / 125.0f);
  };

  const float inv_s3  = 0.5773502691896258f;
  const float pw1     = 0.35355339059327373f;          // sqrt(3/24)
  const float pw1_is3 = pw1 * inv_s3;
  const float pw0_is3 = 0.25f * inv_s3;
  const float inv_s8  = 0.35355339059327373f;          // 1/sqrt(8)
  const bool  center  = (t == 62);                     // (2,2,2)

  for (int e = threadIdx.x; e < 1024; e += 128) {
    const int a = e >> 5;      // input channel
    const int b = e & 31;      // output channel
    float val;
    if (a < 8) {
      if (b < 8) {                                      // K00 (sh0 == 1)
        val = 0.25f * Wcol(a * 8 + b);
        if (center) val += inv_s8 * wsc0[a * 8 + b];
      } else {                                          // K01
        const int bb = b - 8, w = bb / 3, kc = bb - w * 3;
        val = pw1_is3 * Wcol(64 + a * 8 + w) * sh[1 + kc];
      }
    } else {
      const int aa = a - 8, u = aa / 3, m = aa - u * 3;
      if (b < 8) {                                      // K10
        val = pw0_is3 * Wcol(192 + u * 8 + b) * sh[1 + m];
      } else {                                          // K11
        const int bb = b - 8, w = bb / 3, kc = bb - w * 3;
        float v = Wcol(256 + u * 8 + w) * T[m][kc];
        if (m == kc) v += inv_s3 * Wcol(128 + u * 8 + w);
        val = pw1 * v;
        if (center && m == kc) val += inv_s8 * wsc1[u * 8 + w];
      }
    }
    ks[a][b] = val;
  }
  __syncthreads();

  // swizzle into hi/lo bf16 A-fragment order; wave wv -> frag (mt = wv>>1, pl = wv&1)
  const int lane = threadIdx.x & 31, wv = threadIdx.x >> 5;
  const int half = lane >> 4;
  const int pl   = wv & 1;
  const int o    = (wv >> 1) * 16 + (lane & 15);
  unsigned short* dstf =
      filtb + (size_t)t * TAP_STRIDE + wv * FRAG_STRIDE + lane * LANE_STRIDE;
#pragma unroll
  for (int e = 0; e < 16; ++e) {
    const int K = half * 16 + e;
    const float v = ks[K][o];
    const unsigned short h = f2bf(v);
    dstf[e] = pl ? f2bf(v - bf2f(h)) : h;
  }
#pragma unroll
  for (int e = 16; e < LANE_STRIDE; ++e) dstf[e] = 0;   // keep pads defined
}

// ---------------------------------------------------------------------------
// Kernel 2: conv via split-bf16 WMMA. One block per (b, x, y); 4 waves cover
// z=0..63 (16 voxels each). Per (dx,dy) tap:
//   - filter slice (30 KB, contiguous) -> LDS via global_load_async_to_lds_b128
//   - input row (32ch x 68z, zero-padded) -> hi/lo bf16 planes in LDS
// Per tap: 6 v16bf ds loads + 6 WMMAs (acc0/acc1 interleaved).
// ---------------------------------------------------------------------------
__global__ __launch_bounds__(128)
void conv_wmma(const float* __restrict__ xin,
               const unsigned short* __restrict__ filtb,
               float* __restrict__ out) {
  __shared__ __align__(16) unsigned short ldsA[5 * TAP_STRIDE];  // 30720 B
  __shared__ __align__(16) unsigned short ldsBh[68 * BROW];      //  5440 B
  __shared__ __align__(16) unsigned short ldsBl[68 * BROW];      //  5440 B

  const int blk  = blockIdx.x;              // b*4096 + xx*64 + yy
  const int b    = blk >> 12;
  const int xx   = (blk >> 6) & 63;
  const int yy   = blk & 63;
  const int tid  = threadIdx.x;
  const int lane = tid & 31;
  const int wv   = tid >> 5;
  const int pbase  = wv * 16;               // this wave's voxel base (z)
  const int nlow   = lane & 15;             // N index (voxel within tile)
  const int half16 = (lane >> 4) * 16;      // B-fragment channel base

  v8f acc0 = {0.f, 0.f, 0.f, 0.f, 0.f, 0.f, 0.f, 0.f};  // out ch  0..15
  v8f acc1 = {0.f, 0.f, 0.f, 0.f, 0.f, 0.f, 0.f, 0.f};  // out ch 16..31

  for (int dx = 0; dx < 5; ++dx) {
    const int xi = xx + dx - 2;
    for (int dy = 0; dy < 5; ++dy) {
      const int yi = yy + dy - 2;
      __syncthreads();                       // protect previous tile

      // --- 1) async DMA: contiguous 30 KB filter slice -> ldsA -------------
      {
        const char* srcp = (const char*)(filtb +
                           (size_t)(dx * 25 + dy * 5) * TAP_STRIDE);
#pragma unroll
        for (int i = 0; i < 15; ++i) {       // 5*TAP_STRIDE*2 / 16 / 128 = 15
          const int off = (tid + i * 128) * 16;
          const unsigned loff = (unsigned)(uintptr_t)((const char*)ldsA + off);
          const char*    g    = srcp + off;
          asm volatile("global_load_async_to_lds_b128 %0, %1, off"
                       :: "v"(loff), "v"(g)
                       : "memory");
        }
      }

      // --- 2) input row (zero-padded) -> hi/lo bf16 planes -----------------
      const bool rowok = ((unsigned)xi < 64u) && ((unsigned)yi < 64u);
      const float* src = xin + ((((size_t)b * NCH) * NX + xi) * NX + yi) * NX;
      for (int idx = tid; idx < NCH * 68; idx += 128) {
        const int c  = idx / 68;
        const int zi = idx - c * 68;
        const int z  = zi - 2;
        float v = 0.f;
        if (rowok && (unsigned)z < 64u) v = src[(size_t)c * CHW + z];
        const unsigned short h = f2bf(v);
        ldsBh[zi * BROW + c] = h;
        ldsBl[zi * BROW + c] = f2bf(v - bf2f(h));
      }

#if __has_builtin(__builtin_amdgcn_s_wait_asynccnt)
      __builtin_amdgcn_s_wait_asynccnt(0);
#else
      asm volatile("s_wait_asynccnt 0x0" ::: "memory");
#endif
      __syncthreads();

      // --- 3) 5(dz) taps: 6 WMMAs each, all operands from LDS --------------
#pragma unroll
      for (int dzt = 0; dzt < 5; ++dzt) {
        const int zrow = (pbase + nlow + dzt) * BROW + half16;
        const v16bf bh = *(const v16bf*)&ldsBh[zrow];
        const v16bf bl = *(const v16bf*)&ldsBl[zrow];
        const unsigned short* ab = ldsA + dzt * TAP_STRIDE + lane * LANE_STRIDE;
        const v16bf a0h = *(const v16bf*)(ab);                    // mt0 hi
        const v16bf a0l = *(const v16bf*)(ab + 1 * FRAG_STRIDE);  // mt0 lo
        const v16bf a1h = *(const v16bf*)(ab + 2 * FRAG_STRIDE);  // mt1 hi
        const v16bf a1l = *(const v16bf*)(ab + 3 * FRAG_STRIDE);  // mt1 lo
        acc0 = WMMA_BF16(a0h, bh, acc0);
        acc1 = WMMA_BF16(a1h, bh, acc1);
        acc0 = WMMA_BF16(a0h, bl, acc0);
        acc1 = WMMA_BF16(a1h, bl, acc1);
        acc0 = WMMA_BF16(a0l, bh, acc0);
        acc1 = WMMA_BF16(a1l, bh, acc1);
      }
    }
  }

  // epilogue: D layout -> lane%16 = voxel, VGPR r -> M = r (+8 upper lanes)
  const int p     = pbase + nlow;
  const int rbase = (lane >> 4) * 8;
  float* dst = out + ((size_t)b * NCH) * CHW + ((size_t)xx * NX + yy) * NX + p;
#pragma unroll
  for (int rr = 0; rr < 8; ++rr) {
    const int o0 = rr + rbase;         // tile mt=0
    const int o1 = 16 + rr + rbase;    // tile mt=1
    dst[(size_t)o0 * CHW] = acc0[rr];
    dst[(size_t)o1 * CHW] = acc1[rr];
  }
}

// ---------------------------------------------------------------------------
extern "C" void kernel_launch(void* const* d_in, const int* in_sizes, int n_in,
                              void* d_out, int out_size, void* d_ws, size_t ws_size,
                              hipStream_t stream) {
  const float* x    = (const float*)d_in[0];   // (2, 32, 64, 64, 64)
  const float* wgt  = (const float*)d_in[1];   // (5, 320)
  const float* wsc0 = (const float*)d_in[2];   // (8, 8)
  const float* wsc1 = (const float*)d_in[3];   // (8, 8)
  unsigned short* filtb = (unsigned short*)d_ws;   // needs 750 KB scratch

  build_filter<<<125, 128, 0, stream>>>(wgt, wsc0, wsc1, filtb);
  conv_wmma<<<2 * 64 * 64, 128, 0, stream>>>(x, filtb, (float*)d_out);
}
